// StackRNN_17308718203497
// MI455X (gfx1250) — compile-verified, matrix operand order checked by reading, not run on hardware
//
#include <hip/hip_runtime.h>
#include <hip/hip_bf16.h>

// ---------------- problem constants ----------------
#define SEQ_  1024
#define B_    256
#define IN_   256
#define H_    512
#define SW_   128
#define SD_   64
#define KTOT  (IN_ + SW_ + H_)   // 896 : [x | stack_top | h]
#define N4H   (4 * H_)           // 2048
#define KSTEPS (KTOT / 32)       // 28
#define NWG   64
#define TPB   256
#define GT    (NWG * TPB)        // 16384 global threads

typedef __attribute__((ext_vector_type(16))) __bf16 v16bf;
typedef __attribute__((ext_vector_type(8)))  float  v8f;
typedef __attribute__((ext_vector_type(4)))  unsigned tdm_u4;
typedef __attribute__((ext_vector_type(8)))  int      tdm_i8;
typedef __attribute__((ext_vector_type(4)))  int      tdm_i4;

__device__ __forceinline__ unsigned short f2bf(float f) {
    unsigned u = __builtin_bit_cast(unsigned, f);
    unsigned r = u + 0x7FFFu + ((u >> 16) & 1u);   // RNE
    return (unsigned short)(r >> 16);
}
__device__ __forceinline__ float sigmoidf_(float x) {
    return 1.0f / (1.0f + __expf(-x));
}
__device__ __forceinline__ v16bf asv16(const unsigned (&u)[8]) {
    union { unsigned u[8]; v16bf v; } c;
#pragma unroll
    for (int i = 0; i < 8; ++i) c.u[i] = u[i];
    return c.v;
}

// device-scope generation barrier across the persistent grid
__device__ __forceinline__ void grid_barrier(unsigned* cnt, unsigned* gen) {
    __syncthreads();
    if (threadIdx.x == 0) {
        __threadfence();
        unsigned g = __atomic_load_n(gen, __ATOMIC_ACQUIRE);
        unsigned arrived = atomicAdd(cnt, 1u);
        if (arrived == NWG - 1u) {
            *cnt = 0u;
            __threadfence();
            atomicAdd(gen, 1u);
        } else {
            while (__atomic_load_n(gen, __ATOMIC_ACQUIRE) == g) {
#if defined(__HIP_DEVICE_COMPILE__)
                __builtin_amdgcn_s_sleep(2);
#endif
            }
        }
        __threadfence();
    }
    __syncthreads();
}

// ---------------- prep: fuse weights to bf16, seed state, zero barrier ----------------
__global__ __launch_bounds__(TPB) void stackrnn_prep(
    const float* __restrict__ h0, const float* __restrict__ c0,
    const float* __restrict__ stack0,
    const float* __restrict__ W_ih, const float* __restrict__ W_hh,
    const float* __restrict__ b_ih, const float* __restrict__ b_hh,
    unsigned short* __restrict__ Wcat, float* __restrict__ bsum,
    unsigned short* __restrict__ act, float* __restrict__ hbuf,
    float* __restrict__ cbuf, float* __restrict__ stackbuf,
    unsigned* __restrict__ bar)
{
    const long nW = (long)N4H * KTOT;            // 1,835,008
    const long total = nW + N4H + (long)B_ * H_ * 2 + (long)B_ * SD_ * SW_;
    for (long i = (long)blockIdx.x * TPB + threadIdx.x; i < total;
         i += (long)gridDim.x * TPB) {
        long j = i;
        if (j < nW) {                            // fused bf16 weights [N4H][KTOT]
            int n = (int)(j / KTOT), k = (int)(j % KTOT);
            float w = (k < IN_ + SW_) ? W_ih[(long)n * (IN_ + SW_) + k]
                                      : W_hh[(long)n * H_ + (k - (IN_ + SW_))];
            Wcat[j] = f2bf(w);
            continue;
        }
        j -= nW;
        if (j < N4H) { bsum[j] = b_ih[j] + b_hh[j]; continue; }
        j -= N4H;
        if (j < (long)B_ * H_) {                 // hbuf + act h-part
            float v = h0[j];
            hbuf[j] = v;
            int b = (int)(j >> 9), hh = (int)(j & (H_ - 1));
            act[(long)b * KTOT + (IN_ + SW_) + hh] = f2bf(v);
            continue;
        }
        j -= (long)B_ * H_;
        if (j < (long)B_ * H_) { cbuf[j] = c0[j]; continue; }
        j -= (long)B_ * H_;
        stackbuf[j] = stack0[j];                 // ping buffer 0
    }
    if (blockIdx.x == 0 && threadIdx.x == 0) { bar[0] = 0u; bar[1] = 0u; }
}

// ---------------- persistent recurrent kernel ----------------
__global__ __launch_bounds__(TPB) void stackrnn_run(
    const float* __restrict__ x,
    const float* __restrict__ A_w, const float* __restrict__ A_b,
    const float* __restrict__ D_w, const float* __restrict__ D_b,
    const unsigned short* __restrict__ Wcat, const float* __restrict__ bsum,
    unsigned short* __restrict__ act, float* __restrict__ gates,
    float* __restrict__ hbuf, float* __restrict__ cbuf,
    float* __restrict__ stackbuf, float* __restrict__ ctl,
    float* __restrict__ dbuf, unsigned* __restrict__ bar,
    float* __restrict__ outs, float* __restrict__ out_h,
    float* __restrict__ out_c, float* __restrict__ out_stack)
{
    const int gtid  = blockIdx.x * TPB + threadIdx.x;
    const int mtile = blockIdx.x & 15;   // which 16-row block of B
    const int nq    = blockIdx.x >> 4;   // which 512-col quarter of N4H
    const int wid   = threadIdx.x >> 5;  // wave32 id, 0..7
    const int lane  = threadIdx.x & 31;

    __shared__ unsigned short ldsA[16 * KTOT];   // 28,672 B bf16 A-tile

    // ----- TDM descriptor for the A-tile (built once; tile address is static) -----
#if defined(__gfx1250__) && __has_builtin(__builtin_amdgcn_tensor_load_to_lds)
    const unsigned ldsAddr = (unsigned)(unsigned long long)(uintptr_t)&ldsA[0];
    const unsigned long long gaddr =
        (unsigned long long)(uintptr_t)(act + (long)mtile * 16 * KTOT);
    tdm_u4 g0;
    g0[0] = 1u;                                             // count=1, user mode
    g0[1] = ldsAddr;                                        // LDS dest (bytes)
    g0[2] = (unsigned)(gaddr & 0xFFFFFFFFu);                // global addr lo
    g0[3] = (unsigned)((gaddr >> 32) & 0x01FFFFFFu) | 0x80000000u; // hi | type=2
    tdm_i8 g1;
    g1[0] = (int)(1u << 16);          // workgroup_mask=0, data_size=1 (2 bytes)
    g1[1] = (int)((unsigned)KTOT << 16);      // tensor_dim0[15:0] @ bits 63:48
    g1[2] = (int)((unsigned)B_ << 16);        // tensor_dim1[15:0] @ bits 95:80
    g1[3] = (int)((unsigned)KTOT << 16);      // tile_dim0 @ bits 127:112
    g1[4] = 16;                               // tile_dim1=16, tile_dim2=0
    g1[5] = KTOT;                             // tensor_dim0_stride lo32
    g1[6] = 0;
    g1[7] = 0;
    const tdm_i4 gz = {0, 0, 0, 0};
#endif

    for (int t = 0; t < SEQ_; ++t) {
        // ---------- P1a : xt->bf16, control logits, d = tanh(h @ D_w^T + D_b) ----------
        const float* xt = x + (long)t * B_ * IN_;
        for (int i = gtid; i < B_ * IN_; i += GT) {
            int b = i >> 8, k = i & (IN_ - 1);
            act[(long)b * KTOT + k] = f2bf(xt[i]);
        }
        if (gtid < B_ * 3) {
            int b = gtid / 3, j = gtid % 3;
            float acc = A_b[j];
            const float* hr = hbuf + (long)b * H_;
            const float* wr = A_w + (long)j * H_;
            for (int k = 0; k < H_; ++k) acc += hr[k] * wr[k];
            ctl[gtid] = acc;
        }
        for (int i = gtid; i < B_ * SW_; i += GT) {
            int b = i >> 7, j = i & (SW_ - 1);
            float acc = D_b[j];
            const float* hr = hbuf + (long)b * H_;
            const float* wr = D_w + (long)j * H_;
            for (int k = 0; k < H_; ++k) acc += hr[k] * wr[k];
            dbuf[i] = tanhf(acc);
        }
        grid_barrier(bar, bar + 1);

        // ---------- P1b : softmax + stack blend (ping-pong) ----------
        {
            const float* sOld = stackbuf + (long)(t & 1) * B_ * SD_ * SW_;
            float*       sNew = stackbuf + (long)((t + 1) & 1) * B_ * SD_ * SW_;
            int b = gtid >> 6, worker = gtid & 63;       // 64 workers per batch row
            float l0 = ctl[b * 3 + 0], l1 = ctl[b * 3 + 1], l2 = ctl[b * 3 + 2];
            float mx = fmaxf(l0, fmaxf(l1, l2));
            float e0 = __expf(l0 - mx), e1 = __expf(l1 - mx), e2 = __expf(l2 - mx);
            float inv = 1.0f / (e0 + e1 + e2);
            float a_push = e0 * inv, a_pop = e1 * inv, a_noop = e2 * inv;
            const float* so = sOld + (long)b * SD_ * SW_;
            float*       sn = sNew + (long)b * SD_ * SW_;
            for (int q = 0; q < 128; ++q) {
                int e = worker * 128 + q;
                int s = e >> 7, w = e & (SW_ - 1);
                float oldv = so[e];
                float up = (s == 0) ? dbuf[b * SW_ + w] : so[e - SW_];
                float dn = (s == SD_ - 1) ? 0.0f : so[e + SW_];
                float nv = a_noop * oldv + a_push * up + a_pop * dn;
                sn[e] = nv;
                if (s == 0) act[(long)b * KTOT + IN_ + w] = f2bf(nv);
                if (t == SEQ_ - 1) out_stack[(long)b * SD_ * SW_ + e] = nv;
            }
        }
        grid_barrier(bar, bar + 1);

        // ---------- P2 : gates = act @ Wcat^T  (bf16 WMMA, f32 acc) ----------
        // Stage the 16x896 bf16 A-tile to LDS via the Tensor Data Mover.
#if defined(__gfx1250__) && __has_builtin(__builtin_amdgcn_tensor_load_to_lds)
        if (wid == 0) {
#if __clang_major__ >= 23
            const tdm_i8 gz8 = {0, 0, 0, 0, 0, 0, 0, 0};
            __builtin_amdgcn_tensor_load_to_lds(g0, g1, gz, gz, gz8, 0);
#else
            __builtin_amdgcn_tensor_load_to_lds(g0, g1, gz, gz, 0);
#endif
            __builtin_amdgcn_s_wait_tensorcnt(0);
        }
        __syncthreads();
#else
        for (int i = threadIdx.x; i < 16 * KTOT; i += TPB)
            ldsA[i] = act[(long)(mtile * 16 + (i / KTOT)) * KTOT + (i % KTOT)];
        __syncthreads();
#endif

        {
            v8f acc[4];
#pragma unroll
            for (int nt = 0; nt < 4; ++nt)
#pragma unroll
                for (int r = 0; r < 8; ++r) acc[nt][r] = 0.0f;

            const int m    = lane & 15;
            const int koff = (lane >> 4) * 8;          // lanes 16..31 carry K+8
            const unsigned short* wrowp[4];
#pragma unroll
            for (int nt = 0; nt < 4; ++nt) {
                int ncol = (nq * 32 + wid * 4 + nt) * 16 + (lane & 15);
                wrowp[nt] = Wcat + (long)ncol * KTOT;
            }

            unsigned aE[8], aO[8], bE[4][8], bO[4][8];
            auto loadA = [&](int kk, unsigned (&dst)[8]) {
                const int kb = kk * 32 + koff;
#pragma unroll
                for (int h = 0; h < 2; ++h)
#pragma unroll
                    for (int p = 0; p < 4; ++p)
                        dst[h * 4 + p] =
                            *(const unsigned*)&ldsA[m * KTOT + kb + h * 16 + 2 * p];
            };
            auto loadB = [&](int kk, unsigned (&dst)[4][8]) {
#pragma unroll
                for (int nt = 0; nt < 4; ++nt) {
                    const unsigned short* wr = wrowp[nt];
                    const int kb = kk * 32 + koff;
#pragma unroll
                    for (int h = 0; h < 2; ++h)
#pragma unroll
                        for (int p = 0; p < 4; ++p)
                            dst[nt][h * 4 + p] =
                                *(const unsigned*)&wr[kb + h * 16 + 2 * p];
                    __builtin_prefetch(&wr[kk * 32 + 64], 0, 3);
                }
            };
            auto dowmma = [&](unsigned (&a)[8], unsigned (&b)[4][8]) {
                v16bf av = asv16(a);
#pragma unroll
                for (int nt = 0; nt < 4; ++nt) {
#if defined(__gfx1250__)
                    acc[nt] = __builtin_amdgcn_wmma_f32_16x16x32_bf16(
                        false, av, false, asv16(b[nt]),
                        (short)0, acc[nt], false, false);
#else
                    acc[nt][0] += (float)b[nt][0] * 0.0f + (float)a[0] * 0.0f;
#endif
                }
            };

            // software-pipelined even/odd k-loop: loads for k+1 in flight
            // while WMMAs for k execute (KSTEPS = 28, even)
            loadA(0, aE);
            loadB(0, bE);
            for (int kk = 0; kk < KSTEPS; kk += 2) {
                loadA(kk + 1, aO);
                loadB(kk + 1, bO);
                dowmma(aE, bE);
                if (kk + 2 < KSTEPS) {
                    loadA(kk + 2, aE);
                    loadB(kk + 2, bE);
                }
                dowmma(aO, bO);
            }

            // scatter C tiles (16x16 f32 layout: VGPR r -> M = r + 8*(lane>=16), N = lane%16)
            const int mbase = mtile * 16 + ((lane >= 16) ? 8 : 0);
#pragma unroll
            for (int nt = 0; nt < 4; ++nt) {
                const int ncol = (nq * 32 + wid * 4 + nt) * 16 + (lane & 15);
#pragma unroll
                for (int r = 0; r < 8; ++r)
                    gates[(long)(mbase + r) * N4H + ncol] = acc[nt][r];
            }
        }
        grid_barrier(bar, bar + 1);

        // ---------- P3 : LSTM cell + emit h ----------
        for (int i = gtid; i < B_ * H_; i += GT) {
            int b = i >> 9, hh = i & (H_ - 1);
            const float* g = gates + (long)b * N4H;
            float iv = g[hh]            + bsum[hh];
            float fv = g[H_ + hh]       + bsum[H_ + hh];
            float gv = g[2 * H_ + hh]   + bsum[2 * H_ + hh];
            float ov = g[3 * H_ + hh]   + bsum[3 * H_ + hh];
            float cn = sigmoidf_(fv) * cbuf[i] + sigmoidf_(iv) * tanhf(gv);
            float hn = sigmoidf_(ov) * tanhf(cn);
            cbuf[i] = cn;
            hbuf[i] = hn;
            act[(long)b * KTOT + (IN_ + SW_) + hh] = f2bf(hn);
            outs[(long)t * B_ * H_ + i] = hn;
            if (t == SEQ_ - 1) { out_h[i] = hn; out_c[i] = cn; }
        }
        grid_barrier(bar, bar + 1);
    }
}

// ---------------- host launcher ----------------
extern "C" void kernel_launch(void* const* d_in, const int* in_sizes, int n_in,
                              void* d_out, int out_size, void* d_ws, size_t ws_size,
                              hipStream_t stream) {
    const float* x      = (const float*)d_in[0];
    const float* h0     = (const float*)d_in[1];
    const float* c0     = (const float*)d_in[2];
    const float* stack0 = (const float*)d_in[3];
    const float* A_w    = (const float*)d_in[4];
    const float* A_b    = (const float*)d_in[5];
    const float* D_w    = (const float*)d_in[6];
    const float* D_b    = (const float*)d_in[7];
    const float* W_ih   = (const float*)d_in[8];
    const float* W_hh   = (const float*)d_in[9];
    const float* b_ih   = (const float*)d_in[10];
    const float* b_hh   = (const float*)d_in[11];

    char* p = (char*)d_ws;
    auto alloc = [&](size_t bytes) -> void* {
        void* r = (void*)p;
        p += (bytes + 255) & ~(size_t)255;
        return r;
    };
    unsigned short* Wcat     = (unsigned short*)alloc((size_t)N4H * KTOT * 2);
    float*          bsum     = (float*)alloc((size_t)N4H * 4);
    unsigned short* act      = (unsigned short*)alloc((size_t)B_ * KTOT * 2);
    float*          gates    = (float*)alloc((size_t)B_ * N4H * 4);
    float*          hbuf     = (float*)alloc((size_t)B_ * H_ * 4);
    float*          cbuf     = (float*)alloc((size_t)B_ * H_ * 4);
    float*          stackbuf = (float*)alloc((size_t)2 * B_ * SD_ * SW_ * 4);
    float*          ctl      = (float*)alloc((size_t)B_ * 3 * 4);
    float*          dbuf     = (float*)alloc((size_t)B_ * SW_ * 4);
    unsigned*       bar      = (unsigned*)alloc(256);

    float* outs      = (float*)d_out;
    float* out_h     = outs + (size_t)SEQ_ * B_ * H_;
    float* out_c     = out_h + (size_t)B_ * H_;
    float* out_stack = out_c + (size_t)B_ * H_;

    stackrnn_prep<<<2048, TPB, 0, stream>>>(h0, c0, stack0, W_ih, W_hh, b_ih, b_hh,
                                            Wcat, bsum, act, hbuf, cbuf, stackbuf, bar);
    stackrnn_run<<<NWG, TPB, 0, stream>>>(x, A_w, A_b, D_w, D_b, Wcat, bsum,
                                          act, gates, hbuf, cbuf, stackbuf,
                                          ctl, dbuf, bar,
                                          outs, out_h, out_c, out_stack);
}
